// CPMMoE_85203561218083
// MI455X (gfx1250) — matrix operand order repaired and checked by reading, not compiled, not run on hardware
//
#include <hip/hip_runtime.h>
#include <hip/hip_bf16.h>

// ---------------------------------------------------------------------------
// Problem constants (from reference): B=8, T=1024, K=1024, O=1024, N=32, R=512
// ---------------------------------------------------------------------------
#define M_ROWS   8192      // B*T
#define KDIM     1024
#define NDIM     32
#define ODIM     1024
#define RDIM     512
#define NCH      64        // 2 branches * 32 channels

#define STRIP    32        // rows per fused-kernel workgroup
#define RANK_LDS_STRIDE 520   // 512 + 8 pad: spreads LDS banks for b128 frag reads

typedef __attribute__((ext_vector_type(16))) __bf16 v16bf;
typedef __attribute__((ext_vector_type(8)))  __bf16 v8bf;
typedef __attribute__((ext_vector_type(8)))  float  v8f;

// ---------------------------------------------------------------------------
// Helpers
// ---------------------------------------------------------------------------
__device__ __forceinline__ unsigned short f2bf(float f) {
    unsigned int u = __float_as_uint(f);
    unsigned int r = u + 0x7FFFu + ((u >> 16) & 1u);   // round-to-nearest-even
    return (unsigned short)(r >> 16);
}

// A fragment: 16x32 bf16, row-major source with stride lda (global or LDS).
// ISA layout: lanes 0-15 hold M=0..15 with K chunks {0..7, 16..23};
//             lanes 16-31 hold M=0..15 with K chunks {8..15, 24..31}.
__device__ __forceinline__ v16bf load_a_frag(const unsigned short* src, int lda,
                                             int row0, int k0, int lane) {
    int r    = lane & 15;
    int koff = (lane >> 4) << 3;               // 0 or 8
    const unsigned short* base = src + (size_t)(row0 + r) * lda + k0 + koff;
    union { v16bf v; v8bf h[2]; } u;
    u.h[0] = *reinterpret_cast<const v8bf*>(base);        // K = k0+koff .. +7
    u.h[1] = *reinterpret_cast<const v8bf*>(base + 16);   // K = k0+16+koff .. +7
    return u.v;
}

// B fragment: 32x16 bf16 from a pre-transposed (N x K row-major) buffer.
// ISA layout: lanes 0-15 hold N=lane with K chunks {0..7, 16..23}; lanes 16-31 N=lane-16, K chunks {8..15, 24..31}.
__device__ __forceinline__ v16bf load_b_frag(const unsigned short* bt, int ldbt,
                                             int n0, int k0, int lane) {
    int n    = lane & 15;
    int koff = (lane >> 4) << 3;               // 0 or 8
    const unsigned short* base = bt + (size_t)(n0 + n) * ldbt + k0 + koff;
    union { v16bf v; v8bf h[2]; } u;
    u.h[0] = reinterpret_cast<const v8bf*>(base)[0];
    u.h[1] = reinterpret_cast<const v8bf*>(base + 16)[0];
    return u.v;
}

__device__ __forceinline__ v8f wmma_bf16(v16bf a, v16bf b, v8f c) {
    return __builtin_amdgcn_wmma_f32_16x16x32_bf16(false, a, false, b,
                                                   (short)0, c, false, false);
}

// C/D layout: VGPR j, lane l -> M = j + 8*(l>=16), N = l & 15
__device__ __forceinline__ void store_c_f32(float* dst, int ldc, int row0, int col0,
                                            int lane, v8f c) {
    int n  = lane & 15;
    int mo = (lane >> 4) << 3;
    #pragma unroll
    for (int j = 0; j < 8; ++j)
        dst[(size_t)(row0 + mo + j) * ldc + col0 + n] = c[j];
}

// Register-blocked GEMM core: MB x NB tiles of 16x16, K-loop over KTOT.
// Per k-step: MB + NB fragment loads feed MB*NB independent WMMAs.
// A may live in LDS (pointer provenance preserved by inlining -> ds_load_b128).
template<int MB, int NB, int KTOT>
__device__ __forceinline__ void gemm_block(const unsigned short* A, int lda, int row0,
                                           const unsigned short* BT, int ldbt, int col0,
                                           int lane, v8f (&acc)[MB][NB]) {
    for (int kk = 0; kk < KTOT; kk += 32) {
        v16bf af[MB], bfm[NB];
        #pragma unroll
        for (int i = 0; i < MB; ++i)
            af[i] = load_a_frag(A, lda, row0 + 16 * i, kk, lane);
        #pragma unroll
        for (int j = 0; j < NB; ++j)
            bfm[j] = load_b_frag(BT, ldbt, col0 + 16 * j, kk, lane);
        #pragma unroll
        for (int i = 0; i < MB; ++i)
            #pragma unroll
            for (int j = 0; j < NB; ++j)
                acc[i][j] = wmma_bf16(af[i], bfm[j], acc[i][j]);
    }
}

// ---------------------------------------------------------------------------
// Packing kernels (f32 -> bf16, with layout transposes for B operands)
// ---------------------------------------------------------------------------
__global__ void pack_z_kernel(const float* __restrict__ z, unsigned short* __restrict__ zb) {
    size_t i = ((size_t)blockIdx.x * blockDim.x + threadIdx.x) * 4;
    float4 v = *reinterpret_cast<const float4*>(z + i);
    ushort4 o;
    o.x = f2bf(v.x); o.y = f2bf(v.y); o.z = f2bf(v.z); o.w = f2bf(v.w);
    *reinterpret_cast<ushort4*>(zb + i) = o;
}

// projT: (64 x 1024) bf16 ; n<32 -> proj0 column n, else proj1 column n-32
__global__ void pack_projT_kernel(const float* __restrict__ p0, const float* __restrict__ p1,
                                  unsigned short* __restrict__ projT) {
    int i = blockIdx.x * 256 + threadIdx.x;        // 64*1024
    int nn = i >> 10, k = i & 1023;
    float v = (nn < 32) ? p0[(size_t)k * 32 + nn] : p1[(size_t)k * 32 + (nn - 32)];
    projT[i] = f2bf(v);
}

// fac2T: (512 x 1024) bf16 transpose of fac2[0:1024]; bias row kept f32
__global__ void pack_fac2T_kernel(const float* __restrict__ fac2,
                                  unsigned short* __restrict__ fac2T,
                                  float* __restrict__ fac2bias) {
    int i = blockIdx.x * 256 + threadIdx.x;        // 512*1024
    int r = i >> 10, k = i & 1023;
    fac2T[i] = f2bf(fac2[(size_t)k * RDIM + r]);
    if (k == 0) fac2bias[r] = fac2[(size_t)KDIM * RDIM + r];
}

// fac0T/fac1T: (512 x 32) bf16 transposes
__global__ void pack_fac01T_kernel(const float* __restrict__ f0, const float* __restrict__ f1,
                                   unsigned short* __restrict__ f0T, unsigned short* __restrict__ f1T) {
    int i = blockIdx.x * 256 + threadIdx.x;        // 512*32
    int r = i >> 5, n = i & 31;
    f0T[i] = f2bf(f0[(size_t)n * RDIM + r]);
    f1T[i] = f2bf(f1[(size_t)n * RDIM + r]);
}

// fac3: (1024 x 512) row-major is already the BT layout we need -> just convert
__global__ void pack_fac3_kernel(const float* __restrict__ f3, unsigned short* __restrict__ f3b) {
    int i = blockIdx.x * 256 + threadIdx.x;        // 1024*512
    f3b[i] = f2bf(f3[i]);
}

// ---------------------------------------------------------------------------
// GEMM 1: p = z @ [proj0 | proj1]  -> (8192 x 64) f32
// 4x2 tile blocks: 128 Mblocks x 2 Nblocks = 256 waves -> 32 blocks x 8 waves
// ---------------------------------------------------------------------------
__global__ void proj_gemm_kernel(const unsigned short* __restrict__ zb,
                                 const unsigned short* __restrict__ projT,
                                 float* __restrict__ p) {
    int lane = threadIdx.x & 31;
    int t = blockIdx.x * 8 + (threadIdx.x >> 5);   // 0..255
    int row0 = (t >> 1) * 64;
    int col0 = (t & 1) * 32;
    v8f acc[4][2] = {};
    gemm_block<4, 2, KDIM>(zb, KDIM, row0, projT, KDIM, col0, lane, acc);
    #pragma unroll
    for (int i = 0; i < 4; ++i)
        #pragma unroll
        for (int j = 0; j < 2; ++j)
            store_c_f32(p, NCH, row0 + 16 * i, col0 + 16 * j, lane, acc[i][j]);
}

// ---------------------------------------------------------------------------
// BatchNorm statistics: mean / rstd per channel (64 channels)
// ---------------------------------------------------------------------------
__global__ void bn_stats_kernel(const float* __restrict__ p,
                                float* __restrict__ mean, float* __restrict__ rstd) {
    int c = blockIdx.x;
    float s = 0.f, s2 = 0.f;
    for (int row = threadIdx.x; row < M_ROWS; row += 256) {
        float v = p[(size_t)row * NCH + c];
        s += v; s2 += v * v;
    }
    __shared__ float sh[256], sh2[256];
    sh[threadIdx.x] = s; sh2[threadIdx.x] = s2;
    __syncthreads();
    for (int off = 128; off > 0; off >>= 1) {
        if (threadIdx.x < off) {
            sh[threadIdx.x]  += sh[threadIdx.x + off];
            sh2[threadIdx.x] += sh2[threadIdx.x + off];
        }
        __syncthreads();
    }
    if (threadIdx.x == 0) {
        float mu  = sh[0] / (float)M_ROWS;
        float var = sh2[0] / (float)M_ROWS - mu * mu;
        mean[c] = mu;
        rstd[c] = rsqrtf(var + 1e-5f);
    }
}

// ---------------------------------------------------------------------------
// BN + exact 1.5-entmax: one wave32 per (row, branch); lane == channel.
// Bitonic sort across lanes + inclusive scans + ballot support count.
// ---------------------------------------------------------------------------
__global__ void entmax_kernel(const float* __restrict__ p,
                              const float* __restrict__ mean, const float* __restrict__ rstd,
                              unsigned short* __restrict__ a12) {
    int lane = threadIdx.x & 31;
    int g = blockIdx.x * 8 + (threadIdx.x >> 5);   // 0..16383
    int branch = g & 1;
    int row = g >> 1;
    int c = branch * 32 + lane;

    float x = (p[(size_t)row * NCH + c] - mean[c]) * rstd[c];
    x *= 0.5f;
    float m = x;
    #pragma unroll
    for (int o = 16; o > 0; o >>= 1) m = fmaxf(m, __shfl_xor(m, o, 32));
    x -= m;

    // descending bitonic sort of a copy across the 32 lanes
    float v = x;
    #pragma unroll
    for (int k = 2; k <= 32; k <<= 1) {
        #pragma unroll
        for (int j = k >> 1; j > 0; j >>= 1) {
            float o = __shfl_xor(v, j, 32);
            bool up    = ((lane & k) == 0);
            bool lower = ((lane & j) == 0);
            float mn = fminf(v, o), mx = fmaxf(v, o);
            v = (up == lower) ? mx : mn;           // full pass -> descending
        }
    }

    // inclusive scans of xs and xs^2
    float cs = v, cs2 = v * v;
    #pragma unroll
    for (int o = 1; o < 32; o <<= 1) {
        float t  = __shfl_up(cs,  o, 32);
        float t2 = __shfl_up(cs2, o, 32);
        if (lane >= o) { cs += t; cs2 += t2; }
    }
    float kf  = (float)(lane + 1);
    float mk  = cs  / kf;
    float mk2 = cs2 / kf;
    float ss    = kf * (mk2 - mk * mk);
    float delta = (1.0f - ss) / kf;
    float tau   = mk - sqrtf(fmaxf(delta, 0.0f));

    unsigned int support = (unsigned int)__ballot(tau <= v);
    int k_star = __popc(support) - 1;
    float tau_star = __shfl(tau, k_star, 32);

    float pv = fmaxf(x - tau_star, 0.0f);
    a12[(size_t)row * NCH + c] = f2bf(pv * pv);
}

// ---------------------------------------------------------------------------
// Fused rank+out kernel, one 32-row strip per workgroup (256 strips).
//   Phase 0: stage zb strip (32x1024 bf16 = 64 KB) into LDS.
//   Phase 1: rank = (a1@fac0)*(a2@fac1)*(zb@fac2 + bias) -> LDS (32x520 bf16).
//   Phase 2: out strip = rank_strip @ fac3^T -> global f32.
// The 8.4 MB global `rank` buffer and its 32x/64x L2 recirculation vanish;
// zb is read exactly once per strip. LDS: 64KB + 33KB = 97KB per workgroup.
// ---------------------------------------------------------------------------
__global__ void fused_rank_out_kernel(const unsigned short* __restrict__ zb,
                                      const unsigned short* __restrict__ fac2T,
                                      const float* __restrict__ fac2bias,
                                      const unsigned short* __restrict__ a12,
                                      const unsigned short* __restrict__ fac0T,
                                      const unsigned short* __restrict__ fac1T,
                                      const unsigned short* __restrict__ fac3b,
                                      float* __restrict__ out) {
    __shared__ __align__(16) unsigned short sh_zb[STRIP * KDIM];             // 64 KB
    __shared__ __align__(16) unsigned short sh_rank[STRIP * RANK_LDS_STRIDE]; // 33 KB

    int lane = threadIdx.x & 31;
    int wid  = threadIdx.x >> 5;          // 0..7
    int grow0 = blockIdx.x * STRIP;       // strip base row

    // ---- phase 0: cooperative 16B copies of the zb strip into LDS ----
    {
        const ulonglong2* src = reinterpret_cast<const ulonglong2*>(zb + (size_t)grow0 * KDIM);
        ulonglong2*       dst = reinterpret_cast<ulonglong2*>(sh_zb);
        #pragma unroll
        for (int i = 0; i < (STRIP * KDIM / 8) / 256; ++i)
            dst[i * 256 + threadIdx.x] = src[i * 256 + threadIdx.x];
    }
    __syncthreads();

    // ---- phase 1: rank strip into LDS (each wave: 64 cols in 2 passes) ----
    int n  = lane & 15;
    int mo = (lane >> 4) << 3;
    #pragma unroll
    for (int pass = 0; pass < 2; ++pass) {
        int col0 = wid * 64 + pass * 32;                     // 0..511
        v8f m1[2][2] = {}, m2[2][2] = {};
        gemm_block<2, 2, 32>(a12,      NCH, grow0, fac0T, NDIM, col0, lane, m1);
        gemm_block<2, 2, 32>(a12 + 32, NCH, grow0, fac1T, NDIM, col0, lane, m2);
        v8f accF[2][2] = {};
        gemm_block<2, 2, KDIM>(sh_zb, KDIM, 0, fac2T, KDIM, col0, lane, accF);
        #pragma unroll
        for (int j = 0; j < 2; ++j) {
            float bias = fac2bias[col0 + 16 * j + n];
            #pragma unroll
            for (int i = 0; i < 2; ++i) {
                #pragma unroll
                for (int e = 0; e < 8; ++e) {
                    float val = m1[i][j][e] * m2[i][j][e] * (accF[i][j][e] + bias);
                    sh_rank[(16 * i + mo + e) * RANK_LDS_STRIDE + col0 + 16 * j + n] = f2bf(val);
                }
            }
        }
    }
    __syncthreads();

    // ---- phase 2: out strip (each wave: 128 cols in 2 passes of 2x4) ----
    #pragma unroll
    for (int pass = 0; pass < 2; ++pass) {
        int col0 = wid * 128 + pass * 64;                    // 0..1023
        v8f acc[2][4] = {};
        gemm_block<2, 4, RDIM>(sh_rank, RANK_LDS_STRIDE, 0, fac3b, RDIM, col0, lane, acc);
        #pragma unroll
        for (int i = 0; i < 2; ++i)
            #pragma unroll
            for (int j = 0; j < 4; ++j)
                store_c_f32(out, ODIM, grow0 + 16 * i, col0 + 16 * j, lane, acc[i][j]);
    }
}

// ---------------------------------------------------------------------------
// Workspace layout (bytes, all 256-aligned)
// ---------------------------------------------------------------------------
#define OFF_ZB       ((size_t)0)               // u16 8192*1024   = 16777216 B
#define OFF_PROJT    ((size_t)16777216)        // u16 64*1024     =   131072 B
#define OFF_FAC2T    ((size_t)16908288)        // u16 512*1024    =  1048576 B
#define OFF_FAC2B    ((size_t)17956864)        // f32 512         =     2048 B
#define OFF_FAC0T    ((size_t)17958912)        // u16 512*32      =    32768 B
#define OFF_FAC1T    ((size_t)17991680)        // u16 512*32      =    32768 B
#define OFF_FAC3B    ((size_t)18024448)        // u16 1024*512    =  1048576 B
#define OFF_P        ((size_t)19073024)        // f32 8192*64     =  2097152 B
#define OFF_MEAN     ((size_t)21170176)        // f32 64          (pad 256)
#define OFF_RSTD     ((size_t)21170432)        // f32 64          (pad 256)
#define OFF_A12      ((size_t)21170688)        // u16 8192*64     =  1048576 B
// total ~22.2 MB

extern "C" void kernel_launch(void* const* d_in, const int* in_sizes, int n_in,
                              void* d_out, int out_size, void* d_ws, size_t ws_size,
                              hipStream_t stream) {
    const float* z     = (const float*)d_in[0];
    const float* proj0 = (const float*)d_in[1];
    const float* proj1 = (const float*)d_in[2];
    const float* fac0  = (const float*)d_in[3];
    const float* fac1  = (const float*)d_in[4];
    const float* fac2  = (const float*)d_in[5];
    const float* fac3  = (const float*)d_in[6];
    float* out = (float*)d_out;

    char* ws = (char*)d_ws;
    unsigned short* zb     = (unsigned short*)(ws + OFF_ZB);
    unsigned short* projT  = (unsigned short*)(ws + OFF_PROJT);
    unsigned short* fac2T  = (unsigned short*)(ws + OFF_FAC2T);
    float*          fac2b  = (float*)(ws + OFF_FAC2B);
    unsigned short* fac0T  = (unsigned short*)(ws + OFF_FAC0T);
    unsigned short* fac1T  = (unsigned short*)(ws + OFF_FAC1T);
    unsigned short* fac3b  = (unsigned short*)(ws + OFF_FAC3B);
    float*          p      = (float*)(ws + OFF_P);
    float*          meanp  = (float*)(ws + OFF_MEAN);
    float*          rstdp  = (float*)(ws + OFF_RSTD);
    unsigned short* a12    = (unsigned short*)(ws + OFF_A12);

    // 1) pack / transpose to bf16
    pack_z_kernel     <<<8192, 256, 0, stream>>>(z, zb);
    pack_projT_kernel <<< 256, 256, 0, stream>>>(proj0, proj1, projT);
    pack_fac2T_kernel <<<2048, 256, 0, stream>>>(fac2, fac2T, fac2b);
    pack_fac01T_kernel<<<  64, 256, 0, stream>>>(fac0, fac1, fac0T, fac1T);
    pack_fac3_kernel  <<<2048, 256, 0, stream>>>(fac3, fac3b);

    // 2) projections (both branches fused, N=64)
    proj_gemm_kernel  <<<  32, 256, 0, stream>>>(zb, projT, p);

    // 3) BN statistics
    bn_stats_kernel   <<<  64, 256, 0, stream>>>(p, meanp, rstdp);

    // 4) BN + entmax1.5 -> a1|a2 (bf16)
    entmax_kernel     <<<2048, 256, 0, stream>>>(p, meanp, rstdp, a12);

    // 5+6) fused: rank strip in LDS, then out = rank @ fac3^T (f32)
    fused_rank_out_kernel<<<M_ROWS / STRIP, 256, 0, stream>>>(
        zb, fac2T, fac2b, a12, fac0T, fac1T, fac3b, out);
}
